// TripletInteraction_78408922956499
// MI455X (gfx1250) — matrix operand order repaired and checked by reading, not compiled, not run on hardware
//
#include <hip/hip_runtime.h>
#include <hip/hip_bf16.h>

#define E_N     100000
#define T3_N    1200000
#define KMAX_N  12
#define S_N     7
#define DEDGE   512
#define DTRIP   64
#define DBIL    64
#define DRBF    16
#define DCBF    16
#define INV_SQRT2 0.7071067811865476f

typedef __bf16 bf16_t;
typedef __attribute__((ext_vector_type(16))) __bf16 v16bf;
typedef __attribute__((ext_vector_type(8)))  __bf16 v8bf;
typedef __attribute__((ext_vector_type(4)))  __bf16 v4bf;
typedef __attribute__((ext_vector_type(8)))  float  v8f;

__device__ __forceinline__ float silu_f(float x) { return x / (1.0f + __expf(-x)); }

// A fragment (16x32 bf16, row-major LDS tile, row stride `ld` halves).
// Lane l holds row (l&15); K base 0 (lanes 0-15) / 8 (lanes 16-31);
// regs 0-3 cover K=kb..kb+7, regs 4-7 cover K=16+kb..16+kb+7 -> two ds_load_b128.
__device__ __forceinline__ v16bf frag_a(const bf16_t* p, int ld) {
    const int l  = threadIdx.x & 31;
    const bf16_t* q = p + (l & 15) * ld + ((l < 16) ? 0 : 8);
    v8bf lo = *(const v8bf*)(q);
    v8bf hi = *(const v8bf*)(q + 16);
    return __builtin_shufflevector(lo, hi, 0,1,2,3,4,5,6,7,8,9,10,11,12,13,14,15);
}

// B fragment (32x16 bf16) from a column-major LDS tile (Bt[col][k], stride `ld`).
// Lane l holds col (l&15); K base 0 (lanes 0-15) / 8 (lanes 16-31); regs 0-3 K=kb..kb+7, regs 4-7 K=16+kb..16+kb+7.
__device__ __forceinline__ v16bf frag_b(const bf16_t* p, int ld) {
    const int l  = threadIdx.x & 31;
    const bf16_t* q = p + (l & 15) * ld + ((l < 16) ? 0 : 8);
    v8bf lo = *(const v8bf*)(q);
    v8bf hi = *(const v8bf*)(q + 16);
    return __builtin_shufflevector(lo, hi, 0,1,2,3,4,5,6,7,8,9,10,11,12,13,14,15);
}

__device__ __forceinline__ v4bf pack4(float4 v) {
    v4bf p;
    p[0] = (bf16_t)v.x; p[1] = (bf16_t)v.y; p[2] = (bf16_t)v.z; p[3] = (bf16_t)v.w;
    return p;
}

// ---------------- K1: x_ba = silu(m @ W_ba) * (rbf3 @ W_rbf)  -> bf16 (E,512)
__global__ __launch_bounds__(256) void k1_dense_ba(
    const float* __restrict__ m, const float* __restrict__ rbf3,
    const float* __restrict__ W_ba, const float* __restrict__ W_rbf,
    bf16_t* __restrict__ x_ba)
{
    __shared__ __align__(16) bf16_t As[128 * 32];
    __shared__ __align__(16) bf16_t Bs[128 * 32];
    const int rowBase = blockIdx.x * 128;
    const int colBase = blockIdx.y * 128;
    const int tid = threadIdx.x;
    const int w   = tid >> 5;

    v8f acc[8]  = {};
    v8f acc2[8] = {};

    for (int k0 = 0; k0 < DEDGE; k0 += 32) {
        __syncthreads();
        #pragma unroll
        for (int i = 0; i < 4; ++i) {               // A: 128x32 f32 as float4
            int idx4 = tid + i * 256;
            int r = idx4 >> 3, k = (idx4 & 7) * 4, gr = rowBase + r;
            float4 v = make_float4(0.f, 0.f, 0.f, 0.f);
            if (gr < E_N) v = *(const float4*)(m + (size_t)gr * DEDGE + k0 + k);
            *(v4bf*)(&As[r * 32 + k]) = pack4(v);
        }
        #pragma unroll
        for (int i = 0; i < 4; ++i) {               // B: 32x128 f32 as float4 -> col-major
            int idx4 = tid + i * 256;
            int k = idx4 >> 5, c = (idx4 & 31) * 4;
            float4 v = *(const float4*)(W_ba + (size_t)(k0 + k) * DEDGE + colBase + c);
            Bs[(c + 0) * 32 + k] = (bf16_t)v.x;
            Bs[(c + 1) * 32 + k] = (bf16_t)v.y;
            Bs[(c + 2) * 32 + k] = (bf16_t)v.z;
            Bs[(c + 3) * 32 + k] = (bf16_t)v.w;
        }
        __syncthreads();
        v16bf a = frag_a(As + (w * 16) * 32, 32);
        v16bf bfr[8];
        #pragma unroll
        for (int ct = 0; ct < 8; ++ct) bfr[ct] = frag_b(Bs + (ct * 16) * 32, 32);
        #pragma unroll
        for (int ct = 0; ct < 8; ++ct)
            acc[ct] = __builtin_amdgcn_wmma_f32_16x16x32_bf16(
                false, a, false, bfr[ct], (short)0, acc[ct], false, false);
    }

    // rbf projection: one K-step, K=16 zero-padded to 32
    __syncthreads();
    #pragma unroll
    for (int i = 0; i < 4; ++i) {
        int idx4 = tid + i * 256;
        int r = idx4 >> 3, k = (idx4 & 7) * 4, gr = rowBase + r;
        float4 v = make_float4(0.f, 0.f, 0.f, 0.f);
        if (k < DRBF && gr < E_N) v = *(const float4*)(rbf3 + (size_t)gr * DRBF + k);
        *(v4bf*)(&As[r * 32 + k]) = pack4(v);
    }
    #pragma unroll
    for (int i = 0; i < 4; ++i) {
        int idx4 = tid + i * 256;
        int k = idx4 >> 5, c = (idx4 & 31) * 4;
        float4 v = make_float4(0.f, 0.f, 0.f, 0.f);
        if (k < DRBF) v = *(const float4*)(W_rbf + (size_t)k * DEDGE + colBase + c);
        Bs[(c + 0) * 32 + k] = (bf16_t)v.x;
        Bs[(c + 1) * 32 + k] = (bf16_t)v.y;
        Bs[(c + 2) * 32 + k] = (bf16_t)v.z;
        Bs[(c + 3) * 32 + k] = (bf16_t)v.w;
    }
    __syncthreads();
    {
        v16bf a = frag_a(As + (w * 16) * 32, 32);
        v16bf bfr[8];
        #pragma unroll
        for (int ct = 0; ct < 8; ++ct) bfr[ct] = frag_b(Bs + (ct * 16) * 32, 32);
        #pragma unroll
        for (int ct = 0; ct < 8; ++ct)
            acc2[ct] = __builtin_amdgcn_wmma_f32_16x16x32_bf16(
                false, a, false, bfr[ct], (short)0, acc2[ct], false, false);
    }

    const int l    = tid & 31;
    const int rOff = w * 16 + ((l < 16) ? 0 : 8);
    #pragma unroll
    for (int ct = 0; ct < 8; ++ct) {
        int gc = colBase + ct * 16 + (l & 15);
        #pragma unroll
        for (int v = 0; v < 8; ++v) {
            int gr = rowBase + rOff + v;
            if (gr < E_N)
                x_ba[(size_t)gr * DEDGE + gc] = (bf16_t)(silu_f(acc[ct][v]) * acc2[ct][v]);
        }
    }
}

// ---------------- K2: xdown = silu(x_ba @ W_down) -> bf16 (E,64)
__global__ __launch_bounds__(256) void k2_down(
    const bf16_t* __restrict__ x_ba, const float* __restrict__ W_down,
    bf16_t* __restrict__ xdown)
{
    __shared__ __align__(16) bf16_t As[128 * 32];
    __shared__ __align__(16) bf16_t Bs[64 * 32];
    const int rowBase = blockIdx.x * 128;
    const int tid = threadIdx.x;
    const int w   = tid >> 5;

    v8f acc[4] = {};
    for (int k0 = 0; k0 < DEDGE; k0 += 32) {
        __syncthreads();
        #pragma unroll
        for (int i = 0; i < 2; ++i) {               // A: 128x32 bf16, uint4 = 8 halves
            int idx4 = tid + i * 256;
            int r = idx4 >> 2, k = (idx4 & 3) * 8, gr = rowBase + r;
            uint4 v = make_uint4(0u, 0u, 0u, 0u);
            if (gr < E_N) v = *(const uint4*)(x_ba + (size_t)gr * DEDGE + k0 + k);
            *(uint4*)(&As[r * 32 + k]) = v;
        }
        #pragma unroll
        for (int i = 0; i < 2; ++i) {               // B: 32x64 f32 as float4 -> col-major
            int idx4 = tid + i * 256;
            int k = idx4 >> 4, c = (idx4 & 15) * 4;
            float4 v = *(const float4*)(W_down + (size_t)(k0 + k) * DTRIP + c);
            Bs[(c + 0) * 32 + k] = (bf16_t)v.x;
            Bs[(c + 1) * 32 + k] = (bf16_t)v.y;
            Bs[(c + 2) * 32 + k] = (bf16_t)v.z;
            Bs[(c + 3) * 32 + k] = (bf16_t)v.w;
        }
        __syncthreads();
        v16bf a = frag_a(As + (w * 16) * 32, 32);
        v16bf bfr[4];
        #pragma unroll
        for (int ct = 0; ct < 4; ++ct) bfr[ct] = frag_b(Bs + (ct * 16) * 32, 32);
        #pragma unroll
        for (int ct = 0; ct < 4; ++ct)
            acc[ct] = __builtin_amdgcn_wmma_f32_16x16x32_bf16(
                false, a, false, bfr[ct], (short)0, acc[ct], false, false);
    }
    const int l    = tid & 31;
    const int rOff = w * 16 + ((l < 16) ? 0 : 8);
    #pragma unroll
    for (int ct = 0; ct < 4; ++ct) {
        int gc = ct * 16 + (l & 15);
        #pragma unroll
        for (int v = 0; v < 8; ++v) {
            int gr = rowBase + rOff + v;
            if (gr < E_N)
                xdown[(size_t)gr * DTRIP + gc] = (bf16_t)silu_f(acc[ct][v]);
        }
    }
}

// ---------------- K3: m2[id3_reduce_ca, Kidx3, :] = xdown[id3_expand_ba, :]
__global__ __launch_bounds__(256) void k3_scatter(
    const bf16_t* __restrict__ xdown, const int* __restrict__ id3_reduce_ca,
    const int* __restrict__ Kidx3, const int* __restrict__ id3_expand_ba,
    bf16_t* __restrict__ m2)
{
    int tid = blockIdx.x * 256 + threadIdx.x;
    if (tid >= T3_N * 8) return;
    int t = tid >> 3, part = tid & 7;
    int e = id3_reduce_ca[t];
    int k = Kidx3[t];
    int s = id3_expand_ba[t];
    const uint4* src = (const uint4*)(xdown + (size_t)s * DTRIP);
    uint4* dst = (uint4*)(m2 + ((size_t)e * KMAX_N + k) * DTRIP);
    dst[part] = src[part];
}

// ---------------- K4: per-edge einsums + bilinear GEMM -> x (E,64) f32
__global__ __launch_bounds__(128) void k4_bilinear(
    const bf16_t* __restrict__ m2, const float* __restrict__ cbf3_sph,
    const float* __restrict__ cbf3_rbf_W1, const float* __restrict__ W_bil,
    float* __restrict__ x)
{
    __shared__ __align__(16) float  sph[16][KMAX_N][S_N];
    __shared__ __align__(16) float  cw1[16][DCBF][S_N];
    __shared__ __align__(16) bf16_t m2s[16][KMAX_N][DTRIP];
    __shared__ __align__(16) float  sumk[16][S_N][DTRIP];
    __shared__ __align__(16) bf16_t rs[16][DCBF * DTRIP];   // A matrix (16 x 1024)
    __shared__ __align__(16) bf16_t bilB[DBIL * 32];        // col-major B chunk
    const int eBase = blockIdx.x * 16;
    const int tid = threadIdx.x;
    const int w   = tid >> 5;

    for (int idx = tid; idx < 16 * KMAX_N * S_N; idx += 128) {
        int e = idx / (KMAX_N * S_N), r = idx % (KMAX_N * S_N), ge = eBase + e;
        ((float*)sph)[idx] = (ge < E_N) ? cbf3_sph[(size_t)ge * KMAX_N * S_N + r] : 0.0f;
    }
    for (int idx = tid; idx < 16 * DCBF * S_N; idx += 128) {
        int e = idx / (DCBF * S_N), r = idx % (DCBF * S_N), ge = eBase + e;
        ((float*)cw1)[idx] = (ge < E_N) ? cbf3_rbf_W1[(size_t)ge * DCBF * S_N + r] : 0.0f;
    }
    for (int idx4 = tid; idx4 < 16 * KMAX_N * DTRIP / 8; idx4 += 128) { // uint4 = 8 halves
        int e = idx4 / (KMAX_N * DTRIP / 8), r = idx4 % (KMAX_N * DTRIP / 8), ge = eBase + e;
        uint4 v = make_uint4(0u, 0u, 0u, 0u);
        if (ge < E_N) v = ((const uint4*)m2)[(size_t)ge * (KMAX_N * DTRIP / 8) + r];
        ((uint4*)m2s)[idx4] = v;
    }
    __syncthreads();

    for (int idx = tid; idx < 16 * S_N * DTRIP; idx += 128) {   // sum over k=0..11
        int e = idx / (S_N * DTRIP), r = idx % (S_N * DTRIP), s = r >> 6, t = r & 63;
        float a = 0.0f;
        #pragma unroll
        for (int k = 0; k < KMAX_N; ++k) a += sph[e][k][s] * (float)m2s[e][k][t];
        sumk[e][s][t] = a;
    }
    __syncthreads();
    for (int idx = tid; idx < 16 * DCBF * DTRIP; idx += 128) {  // sum over s=0..6
        int e = idx >> 10, r = idx & 1023, c = r >> 6, t = r & 63;
        float a = 0.0f;
        #pragma unroll
        for (int s = 0; s < S_N; ++s) a += cw1[e][c][s] * sumk[e][s][t];
        rs[e][c * DTRIP + t] = (bf16_t)a;
    }

    v8f acc = {};
    for (int ks = 0; ks < 32; ++ks) {               // K = 1024, 32 steps
        __syncthreads();
        #pragma unroll
        for (int i = 0; i < 4; ++i) {               // B chunk: 32x64 f32 as float4
            int idx4 = tid + i * 128;
            int k = idx4 >> 4, o = (idx4 & 15) * 4;
            int j = ks * 32 + k, c = j >> 6, t = j & 63;
            float4 v = *(const float4*)(W_bil + ((size_t)t * DCBF + c) * DBIL + o);
            bilB[(o + 0) * 32 + k] = (bf16_t)v.x;
            bilB[(o + 1) * 32 + k] = (bf16_t)v.y;
            bilB[(o + 2) * 32 + k] = (bf16_t)v.z;
            bilB[(o + 3) * 32 + k] = (bf16_t)v.w;
        }
        __syncthreads();
        v16bf a = frag_a(&rs[0][0] + ks * 32, DCBF * DTRIP);
        v16bf b = frag_b(bilB + (w * 16) * 32, 32);
        acc = __builtin_amdgcn_wmma_f32_16x16x32_bf16(
            false, a, false, b, (short)0, acc, false, false);
    }
    const int l    = tid & 31;
    const int rOff = (l < 16) ? 0 : 8;
    const int gc   = w * 16 + (l & 15);
    #pragma unroll
    for (int v = 0; v < 8; ++v) {
        int ge = eBase + rOff + v;
        if (ge < E_N) x[(size_t)ge * DBIL + gc] = acc[v];
    }
}

// ---------------- K5: x_ca = silu(x @ W_up_ca) -> d_out ; x_ac = silu(x @ W_up_ac) -> ws
__global__ __launch_bounds__(256) void k5_up(
    const float* __restrict__ x, const float* __restrict__ W_up_ca,
    const float* __restrict__ W_up_ac, float* __restrict__ out_ca,
    float* __restrict__ x_ac)
{
    __shared__ __align__(16) bf16_t As[128 * 32];
    __shared__ __align__(16) bf16_t Bs[64 * 32];
    const int rowBase = blockIdx.x * 128;
    const int which   = blockIdx.y >> 3;            // 0: ca, 1: ac
    const int colBase = (blockIdx.y & 7) * 64;
    const float* W = which ? W_up_ac : W_up_ca;
    float* dst     = which ? x_ac : out_ca;
    const int tid = threadIdx.x;
    const int w   = tid >> 5;

    v8f acc[4] = {};
    for (int k0 = 0; k0 < DBIL; k0 += 32) {
        __syncthreads();
        #pragma unroll
        for (int i = 0; i < 4; ++i) {               // A: 128x32 f32 as float4
            int idx4 = tid + i * 256;
            int r = idx4 >> 3, k = (idx4 & 7) * 4, gr = rowBase + r;
            float4 v = make_float4(0.f, 0.f, 0.f, 0.f);
            if (gr < E_N) v = *(const float4*)(x + (size_t)gr * DBIL + k0 + k);
            *(v4bf*)(&As[r * 32 + k]) = pack4(v);
        }
        #pragma unroll
        for (int i = 0; i < 2; ++i) {               // B: 32x64 f32 as float4 -> col-major
            int idx4 = tid + i * 256;
            int k = idx4 >> 4, c = (idx4 & 15) * 4;
            float4 v = *(const float4*)(W + (size_t)(k0 + k) * DEDGE + colBase + c);
            Bs[(c + 0) * 32 + k] = (bf16_t)v.x;
            Bs[(c + 1) * 32 + k] = (bf16_t)v.y;
            Bs[(c + 2) * 32 + k] = (bf16_t)v.z;
            Bs[(c + 3) * 32 + k] = (bf16_t)v.w;
        }
        __syncthreads();
        v16bf a = frag_a(As + (w * 16) * 32, 32);
        v16bf bfr[4];
        #pragma unroll
        for (int ct = 0; ct < 4; ++ct) bfr[ct] = frag_b(Bs + (ct * 16) * 32, 32);
        #pragma unroll
        for (int ct = 0; ct < 4; ++ct)
            acc[ct] = __builtin_amdgcn_wmma_f32_16x16x32_bf16(
                false, a, false, bfr[ct], (short)0, acc[ct], false, false);
    }
    const int l    = tid & 31;
    const int rOff = w * 16 + ((l < 16) ? 0 : 8);
    #pragma unroll
    for (int ct = 0; ct < 4; ++ct) {
        int gc = colBase + ct * 16 + (l & 15);
        #pragma unroll
        for (int v = 0; v < 8; ++v) {
            int gr = rowBase + rOff + v;
            if (gr < E_N)
                dst[(size_t)gr * DEDGE + gc] = silu_f(acc[ct][v]);
        }
    }
}

// ---------------- K6: out = (x_ca + x_ac[id_swap]) * 1/sqrt(2)
__global__ __launch_bounds__(256) void k6_combine(
    const float* __restrict__ x_ac, const int* __restrict__ id_swap,
    float* __restrict__ out)
{
    int tid = blockIdx.x * 256 + threadIdx.x;
    if (tid >= E_N * (DEDGE / 4)) return;
    int e = tid >> 7, q = tid & 127;
    int sw = id_swap[e];
    float4 ca = ((const float4*)out)[(size_t)e * 128 + q];
    float4 ac = ((const float4*)x_ac)[(size_t)sw * 128 + q];
    float4 r;
    r.x = (ca.x + ac.x) * INV_SQRT2;
    r.y = (ca.y + ac.y) * INV_SQRT2;
    r.z = (ca.z + ac.z) * INV_SQRT2;
    r.w = (ca.w + ac.w) * INV_SQRT2;
    ((float4*)out)[(size_t)e * 128 + q] = r;
}

extern "C" void kernel_launch(void* const* d_in, const int* in_sizes, int n_in,
                              void* d_out, int out_size, void* d_ws, size_t ws_size,
                              hipStream_t stream)
{
    const float* m            = (const float*)d_in[0];
    const float* rbf3         = (const float*)d_in[1];
    const float* cbf3_rbf_W1  = (const float*)d_in[2];
    const float* cbf3_sph     = (const float*)d_in[3];
    const float* W_ba         = (const float*)d_in[4];
    const float* W_rbf        = (const float*)d_in[5];
    const float* W_down       = (const float*)d_in[6];
    const float* W_bil        = (const float*)d_in[7];
    const float* W_up_ca      = (const float*)d_in[8];
    const float* W_up_ac      = (const float*)d_in[9];
    const int*   Kidx3        = (const int*)d_in[10];
    const int*   id_swap      = (const int*)d_in[11];
    const int*   id3_expand   = (const int*)d_in[12];
    const int*   id3_reduce   = (const int*)d_in[13];
    float* out = (float*)d_out;

    // Workspace layout (bytes). x_ac aliases [x_ba|xdown|m2] which are dead by K5.
    char* ws = (char*)d_ws;
    const size_t o_xba   = 0;                                   // bf16 E*512  = 102.4 MB
    const size_t o_xdown = o_xba   + (size_t)E_N * DEDGE * 2;   // bf16 E*64   =  12.8 MB
    const size_t o_m2    = o_xdown + (size_t)E_N * DTRIP * 2;   // bf16 E*12*64= 153.6 MB
    const size_t o_x     = o_m2    + (size_t)E_N * KMAX_N * DTRIP * 2; // f32 E*64 = 25.6 MB
    const size_t o_xac   = 0;                                   // f32 E*512 (aliased)

    bf16_t* x_ba  = (bf16_t*)(ws + o_xba);
    bf16_t* xdown = (bf16_t*)(ws + o_xdown);
    bf16_t* m2    = (bf16_t*)(ws + o_m2);
    float*  xbuf  = (float*)(ws + o_x);
    float*  x_ac  = (float*)(ws + o_xac);

    const int rowBlocks = (E_N + 127) / 128;

    k1_dense_ba<<<dim3(rowBlocks, DEDGE / 128), 256, 0, stream>>>(
        m, rbf3, W_ba, W_rbf, x_ba);
    k2_down<<<dim3(rowBlocks), 256, 0, stream>>>(x_ba, W_down, xdown);
    hipMemsetAsync(ws + o_m2, 0, (size_t)E_N * KMAX_N * DTRIP * 2, stream);
    k3_scatter<<<dim3((T3_N * 8 + 255) / 256), 256, 0, stream>>>(
        xdown, id3_reduce, Kidx3, id3_expand, m2);
    k4_bilinear<<<dim3((E_N + 15) / 16), 128, 0, stream>>>(
        m2, cbf3_sph, cbf3_rbf_W1, W_bil, xbuf);
    k5_up<<<dim3(rowBlocks, 16), 256, 0, stream>>>(
        xbuf, W_up_ca, W_up_ac, out, x_ac);
    k6_combine<<<dim3((E_N * (DEDGE / 4) + 255) / 256), 256, 0, stream>>>(
        x_ac, id_swap, out);
}
